// CFGA_59450937311398
// MI455X (gfx1250) — matrix operand order repaired and checked by reading, not compiled, not run on hardware
//
#include <hip/hip_runtime.h>
#include <hip/hip_bf16.h>

// ---------------- problem constants (match reference) ----------------
#define NUSER   10000
#define NITEM   8000
#define NGROUP  4000
#define DIM     64
#define NLAYER  2
#define ATTC    0.5f
#define BATCH   4096
#define NNZ_R   200000
#define NNZ_RUI 300000
#define NNZ_UG  280000
#define NNZ_UI  360000
#define NNZ_GI  240000
#define KSPLIT  4       // K-parallel splits for att GEMM (occupancy)

typedef __attribute__((ext_vector_type(16))) __bf16          v16bf;
typedef __attribute__((ext_vector_type(16))) unsigned short  ushort16;
typedef __attribute__((ext_vector_type(8)))  float           v8f;

__device__ __forceinline__ unsigned short f2bf(float f) {
  unsigned int u = __float_as_uint(f);
  unsigned int r = u + 0x7FFFu + ((u >> 16) & 1u);   // round-to-nearest-even
  return (unsigned short)(r >> 16);
}
__device__ __forceinline__ float bf2f(unsigned short h) {
  return __uint_as_float(((unsigned int)h) << 16);
}

// ---------------- elementwise helpers ----------------
__global__ void cvt_bf16_kernel(const float* __restrict__ src,
                                unsigned short* __restrict__ dst, int n) {
  int i = blockIdx.x * 256 + threadIdx.x;
  if (i < n) dst[i] = f2bf(src[i]);
}
__global__ void add_f32_kernel(float* __restrict__ dst,
                               const float* __restrict__ src, int n) {
  int i = blockIdx.x * 256 + threadIdx.x;
  if (i < n) dst[i] += src[i];
}

// ---- swizzle E[K,64] (f32) into fragment-contiguous bf16 B layout:
//      Esw[(((k0/32)*4 + t)*32 + lane)*16 + hi] = bf16(E[k0 + 16*(lane>>4) + hi][t*16 + lane&15])
__global__ void eswz_kernel(const float* __restrict__ E,
                            unsigned short* __restrict__ Esw, int K) {
  int idx = blockIdx.x * 256 + threadIdx.x;
  if (idx >= K * DIM) return;
  int hi   = idx & 15;
  int lane = (idx >> 4) & 31;
  int t    = (idx >> 9) & 3;
  int kb   = idx >> 11;
  int k = kb * 32 + (lane >> 4) * 16 + hi;
  int n = t * 16 + (lane & 15);
  Esw[idx] = f2bf(E[(size_t)k * DIM + n]);
}

// ---------------- WMMA GEMM:  G[M,N] = X[M,64] * Y[N,64]^T  (bf16 in, bf16 out)
__global__ void __launch_bounds__(32)
syrk_nt_kernel(const unsigned short* __restrict__ X,
               const unsigned short* __restrict__ Y,
               unsigned short* __restrict__ G, int N) {
  const int tm   = blockIdx.x * 16;
  const int tn   = blockIdx.y * 16;
  const int lane = threadIdx.x;          // 0..31
  const int mrow = lane & 15;
  const int hiL  = lane >> 4;            // 0 or 1
  v8f acc = {};
  #pragma unroll
  for (int k0 = 0; k0 < DIM; k0 += 32) {
    ushort16 au, bu;
    const unsigned short* arow = X + (size_t)(tm + mrow) * DIM + k0 + hiL * 16;
    #pragma unroll
    for (int hi = 0; hi < 16; ++hi) au[hi] = arow[hi];
    const unsigned short* brow = Y + (size_t)(tn + mrow) * DIM + k0 + hiL * 16;
    #pragma unroll
    for (int hi = 0; hi < 16; ++hi) bu[hi] = brow[hi];
    v16bf a = __builtin_bit_cast(v16bf, au);
    v16bf b = __builtin_bit_cast(v16bf, bu);
    acc = __builtin_amdgcn_wmma_f32_16x16x32_bf16(false, a, false, b,
                                                  (short)0, acc, false, false);
  }
  #pragma unroll
  for (int v = 0; v < 8; ++v)
    G[(size_t)(tm + v + 8 * hiL) * N + tn + mrow] = f2bf(acc[v]);
}

// ---------------- WMMA GEMM:  Out[M,64] += Att[M,K](bf16) * E(swizzled bf16)
// one wave computes a full 16x64 strip over a strided subset of K-blocks;
// A fragment loaded once per K-step, reused by 4 WMMAs; KSPLIT waves per
// strip commit partial sums with f32 atomics (Out must be pre-zeroed).
__global__ void __launch_bounds__(32)
att_gemm_kernel(const unsigned short* __restrict__ Att,
                const unsigned short* __restrict__ Esw,
                float* __restrict__ Out, int K) {
  const int tm    = blockIdx.x * 16;
  const int split = blockIdx.y;
  const int lane  = threadIdx.x;
  const int mrow  = lane & 15;
  const int hiL   = lane >> 4;
  const int NB    = K >> 5;
  v8f acc0 = {}, acc1 = {}, acc2 = {}, acc3 = {};
  const unsigned short* arow = Att + (size_t)(tm + mrow) * K + hiL * 16;
  for (int kb = split; kb < NB; kb += KSPLIT) {
    const int k0 = kb << 5;
    ushort16 au, b0u, b1u, b2u, b3u;
    #pragma unroll
    for (int hi = 0; hi < 16; ++hi) au[hi] = arow[k0 + hi];
    const unsigned short* bb = Esw + ((size_t)kb * 128 + lane) * 16;
    #pragma unroll
    for (int hi = 0; hi < 16; ++hi) {
      b0u[hi] = bb[hi];
      b1u[hi] = bb[512 + hi];
      b2u[hi] = bb[1024 + hi];
      b3u[hi] = bb[1536 + hi];
    }
    v16bf a = __builtin_bit_cast(v16bf, au);
    acc0 = __builtin_amdgcn_wmma_f32_16x16x32_bf16(false, a, false,
             __builtin_bit_cast(v16bf, b0u), (short)0, acc0, false, false);
    acc1 = __builtin_amdgcn_wmma_f32_16x16x32_bf16(false, a, false,
             __builtin_bit_cast(v16bf, b1u), (short)0, acc1, false, false);
    acc2 = __builtin_amdgcn_wmma_f32_16x16x32_bf16(false, a, false,
             __builtin_bit_cast(v16bf, b2u), (short)0, acc2, false, false);
    acc3 = __builtin_amdgcn_wmma_f32_16x16x32_bf16(false, a, false,
             __builtin_bit_cast(v16bf, b3u), (short)0, acc3, false, false);
  }
  float* orow = Out + (size_t)(tm + 8 * hiL) * DIM + mrow;
  #pragma unroll
  for (int v = 0; v < 8; ++v) {
    atomicAdd(&orow[(size_t)v * DIM +  0], acc0[v]);
    atomicAdd(&orow[(size_t)v * DIM + 16], acc1[v]);
    atomicAdd(&orow[(size_t)v * DIM + 32], acc2[v]);
    atomicAdd(&orow[(size_t)v * DIM + 48], acc3[v]);
  }
}

// ------- row softmax (optionally leaky_relu first), bf16 -> bf16, f32 math ---
__global__ void row_softmax_kernel(const unsigned short* __restrict__ src,
                                   unsigned short* __restrict__ dst,
                                   int cols, int leaky) {
  const size_t row = blockIdx.x;
  const unsigned short* x = src + row * (size_t)cols;
  unsigned short* y = dst + row * (size_t)cols;
  __shared__ float red[256];
  float m = -3.4e38f;
  for (int j = threadIdx.x; j < cols; j += 256) {
    float v = bf2f(x[j]);
    if (leaky && v < 0.f) v *= 0.01f;
    m = fmaxf(m, v);
  }
  red[threadIdx.x] = m; __syncthreads();
  for (int s = 128; s > 0; s >>= 1) {
    if (threadIdx.x < s) red[threadIdx.x] = fmaxf(red[threadIdx.x], red[threadIdx.x + s]);
    __syncthreads();
  }
  m = red[0]; __syncthreads();
  float sum = 0.f;
  for (int j = threadIdx.x; j < cols; j += 256) {
    float v = bf2f(x[j]);
    if (leaky && v < 0.f) v *= 0.01f;
    sum += __expf(v - m);
  }
  red[threadIdx.x] = sum; __syncthreads();
  for (int s = 128; s > 0; s >>= 1) {
    if (threadIdx.x < s) red[threadIdx.x] += red[threadIdx.x + s];
    __syncthreads();
  }
  const float inv = 1.0f / red[0];
  for (int j = threadIdx.x; j < cols; j += 256) {
    float v = bf2f(x[j]);
    if (leaky && v < 0.f) v *= 0.01f;
    y[j] = f2bf(__expf(v - m) * inv);
  }
}

// ---------------- device CSR build (count -> scan -> fill) ----------------
__global__ void count_kernel(const int* __restrict__ rows, int* __restrict__ cnt, int nnz) {
  int e = blockIdx.x * 256 + threadIdx.x;
  if (e < nnz) atomicAdd(&cnt[rows[e]], 1);
}
__global__ void scan_kernel(const int* __restrict__ cnt, int* __restrict__ offs, int n) {
  __shared__ int sh[1024];
  __shared__ int carry;
  if (threadIdx.x == 0) carry = 0;
  __syncthreads();
  for (int base = 0; base < n; base += 1024) {
    int i = base + threadIdx.x;
    int v = (i < n) ? cnt[i] : 0;
    sh[threadIdx.x] = v; __syncthreads();
    for (int s = 1; s < 1024; s <<= 1) {
      int t = (threadIdx.x >= s) ? sh[threadIdx.x - s] : 0;
      __syncthreads();
      sh[threadIdx.x] += t;
      __syncthreads();
    }
    if (i < n) offs[i] = carry + sh[threadIdx.x] - v;   // exclusive
    if (threadIdx.x == 0) carry += sh[1023];
    __syncthreads();
  }
  if (threadIdx.x == 0) offs[n] = carry;
}
__global__ void fill_kernel(const int* __restrict__ rows, const int* __restrict__ cols,
                            const float* __restrict__ vals, const int* __restrict__ offs,
                            int* __restrict__ cursor, int* __restrict__ ccols,
                            float* __restrict__ cvals, int nnz) {
  int e = blockIdx.x * 256 + threadIdx.x;
  if (e >= nnz) return;
  int r = rows[e];
  int p = offs[r] + atomicAdd(&cursor[r], 1);
  ccols[p] = cols[e];
  cvals[p] = vals[e];
}

// ---- CSR logits: out[r, j] = sum_p val[p] * S[col[p], j]  (no atomics, bf16 out)
__global__ void csr_logits_kernel(const int* __restrict__ offs,
                                  const int* __restrict__ ccols,
                                  const float* __restrict__ cvals,
                                  const unsigned short* __restrict__ S,
                                  unsigned short* __restrict__ out, int ncols) {
  const int r = blockIdx.x;
  const int j = blockIdx.y * 256 + threadIdx.x;
  if (j >= ncols) return;
  const int pb = offs[r], pe = offs[r + 1];
  float acc = 0.f;
  for (int p = pb; p < pe; ++p) {
    const unsigned short* srow = S + (size_t)ccols[p] * ncols;
    if (p + 1 < pe)  // prefetch next randomly-gathered sim row chunk
      __builtin_prefetch(S + (size_t)ccols[p + 1] * ncols + j, 0, 0);
    acc += cvals[p] * bf2f(srow[j]);
  }
  out[(size_t)r * ncols + j] = f2bf(acc);
}

// ---------------- narrow (64-wide) atomic SpMM over f32 dense ----------------
__global__ void spmm64_kernel(const int* __restrict__ rows,
                              const int* __restrict__ cols,
                              const float* __restrict__ vals,
                              const float* __restrict__ in,
                              float* __restrict__ out) {
  const int e = blockIdx.x;
  const int j = threadIdx.x;                 // 0..63
  const int r = rows[e], c = cols[e];
  atomicAdd(&out[(size_t)r * DIM + j], vals[e] * in[(size_t)c * DIM + j]);
}

// ---------------- dst = normalize(dst + coef*delta), rows of 64 ----------------
__global__ void axpy_norm_kernel(float* __restrict__ dst,
                                 const float* __restrict__ delta, float coef) {
  const int row = blockIdx.x, j = threadIdx.x;   // 64 threads
  __shared__ float red[64];
  float v = dst[(size_t)row * DIM + j] + coef * delta[(size_t)row * DIM + j];
  red[j] = v * v; __syncthreads();
  for (int s = 32; s > 0; s >>= 1) {
    if (j < s) red[j] += red[j + s];
    __syncthreads();
  }
  float n = fmaxf(sqrtf(red[0]), 1e-12f);
  dst[(size_t)row * DIM + j] = v / n;
}

// ---------------- output gathers ----------------
__global__ void gather_concat_kernel(const int* __restrict__ idx,
                                     const float* __restrict__ A,
                                     const float* __restrict__ Bm,
                                     float* __restrict__ out, float scale) {
  const int i = blockIdx.x, j = threadIdx.x;     // 128 threads
  const int r = idx[i];
  float v = (j < DIM) ? A[(size_t)r * DIM + j] : Bm[(size_t)r * DIM + (j - DIM)];
  out[(size_t)i * (2 * DIM) + j] = v * scale;
}
__global__ void gather_kernel(const int* __restrict__ idx,
                              const float* __restrict__ S,
                              float* __restrict__ out) {
  const int i = blockIdx.x, j = threadIdx.x;     // 64 threads
  out[(size_t)i * DIM + j] = S[(size_t)idx[i] * DIM + j];
}

// =====================================================================
extern "C" void kernel_launch(void* const* d_in, const int* in_sizes, int n_in,
                              void* d_out, int out_size, void* d_ws, size_t ws_size,
                              hipStream_t stream) {
  (void)in_sizes; (void)n_in; (void)out_size; (void)ws_size;
  // ---- inputs ----
  const int*   user_inputs = (const int*)  d_in[0];
  const int*   pos_groups  = (const int*)  d_in[1];
  const int*   neg_groups  = (const int*)  d_in[2];
  const float* user_emb    = (const float*)d_in[3];
  const float* item_emb    = (const float*)d_in[4];
  const float* group_emb   = (const float*)d_in[5];
  const float* ctx_item    = (const float*)d_in[6];
  const float* ctx_group   = (const float*)d_in[7];
  const int*   R_rows  = (const int*)d_in[8];   const int* R_cols  = (const int*)d_in[9];
  const float* R_vals  = (const float*)d_in[10];
  const int*   Ru_rows = (const int*)d_in[11];  const int* Ru_cols = (const int*)d_in[12];
  const float* Ru_vals = (const float*)d_in[13];
  const int*   ug_rows = (const int*)d_in[14];  const int* ug_cols = (const int*)d_in[15];
  const float* ug_vals = (const float*)d_in[16];
  const int*   ui_rows = (const int*)d_in[17];  const int* ui_cols = (const int*)d_in[18];
  const float* ui_vals = (const float*)d_in[19];
  const int*   gi_rows = (const int*)d_in[20];  const int* gi_cols = (const int*)d_in[21];
  const float* gi_vals = (const float*)d_in[22];
  float* out = (float*)d_out;

  // ---- workspace bump allocator ----
  char* ws = (char*)d_ws;
  size_t off = 0;
  auto alloc = [&](size_t bytes) -> void* {
    off = (off + 255) & ~(size_t)255;
    void* p = ws + off;
    off += bytes;
    return p;
  };
  const size_t UD = (size_t)NUSER * DIM, GD = (size_t)NGROUP * DIM, ID = (size_t)NITEM * DIM;

  unsigned short* SIM_G  = (unsigned short*)alloc((size_t)NGROUP * NGROUP * 2);
  unsigned short* SIM_I  = (unsigned short*)alloc((size_t)NITEM  * NITEM  * 2);
  unsigned short* ATT_R  = (unsigned short*)alloc((size_t)NUSER  * NGROUP * 2);
  unsigned short* ATT_RU = (unsigned short*)alloc((size_t)NUSER  * NITEM  * 2);
  unsigned short* SCR = (unsigned short*)alloc((size_t)NUSER * NITEM * 2); // 160MB bf16 scratch
  unsigned short* CTX_G_BF = (unsigned short*)alloc(GD * 2);
  unsigned short* CTX_I_BF = (unsigned short*)alloc(ID * 2);
  unsigned short* ESW      = (unsigned short*)alloc(ID * 2);  // swizzled B operand
  float* UGP   = (float*)alloc(UD * 4);  float* UIP   = (float*)alloc(UD * 4);
  float* GEU   = (float*)alloc(GD * 4);  float* GEI   = (float*)alloc(GD * 4);
  float* IEU   = (float*)alloc(ID * 4);  float* IEG   = (float*)alloc(ID * 4);
  float* UGP_S = (float*)alloc(UD * 4);  float* UIP_S = (float*)alloc(UD * 4);
  float* GEU_S = (float*)alloc(GD * 4);  float* GEI_S = (float*)alloc(GD * 4);
  float* TMP64 = (float*)alloc(UD * 4);
  float* CAT_IN  = (float*)alloc((size_t)(NUSER + NITEM) * DIM * 4);
  float* CAT_OUT = (float*)alloc((size_t)(NUSER + NITEM) * DIM * 4);
  // CSR scratch
  int*   CNT     = (int*)alloc((size_t)NUSER * 4);
  int*   R_OFF   = (int*)alloc((size_t)(NUSER + 1) * 4);
  int*   RU_OFF  = (int*)alloc((size_t)(NUSER + 1) * 4);
  int*   R_CCOL  = (int*)alloc((size_t)NNZ_R * 4);
  float* R_CVAL  = (float*)alloc((size_t)NNZ_R * 4);
  int*   RU_CCOL = (int*)alloc((size_t)NNZ_RUI * 4);
  float* RU_CVAL = (float*)alloc((size_t)NNZ_RUI * 4);

  // ---- phase 0: CSR-ify R and Rui (removes 3.2G f32 atomics from logits) ----
  hipMemsetAsync(CNT, 0, (size_t)NUSER * 4, stream);
  count_kernel<<<(NNZ_R + 255) / 256, 256, 0, stream>>>(R_rows, CNT, NNZ_R);
  scan_kernel<<<1, 1024, 0, stream>>>(CNT, R_OFF, NUSER);
  hipMemsetAsync(CNT, 0, (size_t)NUSER * 4, stream);
  fill_kernel<<<(NNZ_R + 255) / 256, 256, 0, stream>>>(R_rows, R_cols, R_vals,
                                                       R_OFF, CNT, R_CCOL, R_CVAL, NNZ_R);
  hipMemsetAsync(CNT, 0, (size_t)NUSER * 4, stream);
  count_kernel<<<(NNZ_RUI + 255) / 256, 256, 0, stream>>>(Ru_rows, CNT, NNZ_RUI);
  scan_kernel<<<1, 1024, 0, stream>>>(CNT, RU_OFF, NUSER);
  hipMemsetAsync(CNT, 0, (size_t)NUSER * 4, stream);
  fill_kernel<<<(NNZ_RUI + 255) / 256, 256, 0, stream>>>(Ru_rows, Ru_cols, Ru_vals,
                                                         RU_OFF, CNT, RU_CCOL, RU_CVAL, NNZ_RUI);

  // ---- phase 1: sim matrices (once; bf16 end-to-end) ----
  cvt_bf16_kernel<<<(GD + 255) / 256, 256, 0, stream>>>(ctx_group, CTX_G_BF, (int)GD);
  cvt_bf16_kernel<<<(ID + 255) / 256, 256, 0, stream>>>(ctx_item,  CTX_I_BF, (int)ID);

  syrk_nt_kernel<<<dim3(NGROUP / 16, NGROUP / 16), 32, 0, stream>>>(CTX_G_BF, CTX_G_BF, SCR, NGROUP);
  row_softmax_kernel<<<NGROUP, 256, 0, stream>>>(SCR, SIM_G, NGROUP, 0);

  syrk_nt_kernel<<<dim3(NITEM / 16, NITEM / 16), 32, 0, stream>>>(CTX_I_BF, CTX_I_BF, SCR, NITEM);
  row_softmax_kernel<<<NITEM, 256, 0, stream>>>(SCR, SIM_I, NITEM, 0);

  // ---- phase 2: constant attention matrices (computed once, reused both layers) ----
  csr_logits_kernel<<<dim3(NUSER, (NGROUP + 255) / 256), 256, 0, stream>>>(
      R_OFF, R_CCOL, R_CVAL, SIM_G, SCR, NGROUP);
  row_softmax_kernel<<<NUSER, 256, 0, stream>>>(SCR, ATT_R, NGROUP, 1);

  csr_logits_kernel<<<dim3(NUSER, (NITEM + 255) / 256), 256, 0, stream>>>(
      RU_OFF, RU_CCOL, RU_CVAL, SIM_I, SCR, NITEM);
  row_softmax_kernel<<<NUSER, 256, 0, stream>>>(SCR, ATT_RU, NITEM, 1);

  // ---- phase 3: init layer state + part-sum accumulators ----
  hipMemcpyAsync(UGP, user_emb,  UD * 4, hipMemcpyDeviceToDevice, stream);
  hipMemcpyAsync(UIP, user_emb,  UD * 4, hipMemcpyDeviceToDevice, stream);
  hipMemcpyAsync(GEU, group_emb, GD * 4, hipMemcpyDeviceToDevice, stream);
  hipMemcpyAsync(GEI, group_emb, GD * 4, hipMemcpyDeviceToDevice, stream);
  hipMemcpyAsync(IEU, item_emb,  ID * 4, hipMemcpyDeviceToDevice, stream);
  hipMemcpyAsync(IEG, item_emb,  ID * 4, hipMemcpyDeviceToDevice, stream);
  hipMemcpyAsync(UGP_S, user_emb,  UD * 4, hipMemcpyDeviceToDevice, stream);
  hipMemcpyAsync(UIP_S, user_emb,  UD * 4, hipMemcpyDeviceToDevice, stream);
  hipMemcpyAsync(GEU_S, group_emb, GD * 4, hipMemcpyDeviceToDevice, stream);
  hipMemcpyAsync(GEI_S, group_emb, GD * 4, hipMemcpyDeviceToDevice, stream);

  // ---- phase 4: propagation layers ----
  for (int layer = 0; layer < NLAYER; ++layer) {
    // g2u = att_R @ geu ; ugp = normalize(ugp + 0.5*g2u)
    eswz_kernel<<<((int)GD + 255) / 256, 256, 0, stream>>>(GEU, ESW, NGROUP);
    hipMemsetAsync(TMP64, 0, UD * 4, stream);
    att_gemm_kernel<<<dim3(NUSER / 16, KSPLIT), 32, 0, stream>>>(ATT_R, ESW, TMP64, NGROUP);
    axpy_norm_kernel<<<NUSER, 64, 0, stream>>>(UGP, TMP64, ATTC);

    // i2u = att_Rui @ ieu ; uip = normalize(uip + 0.5*i2u)
    eswz_kernel<<<((int)ID + 255) / 256, 256, 0, stream>>>(IEU, ESW, NITEM);
    hipMemsetAsync(TMP64, 0, UD * 4, stream);
    att_gemm_kernel<<<dim3(NUSER / 16, KSPLIT), 32, 0, stream>>>(ATT_RU, ESW, TMP64, NITEM);
    axpy_norm_kernel<<<NUSER, 64, 0, stream>>>(UIP, TMP64, ATTC);

    // ug = spmm(ug, concat(ugp, geu))
    hipMemcpyAsync(CAT_IN,      UGP, UD * 4, hipMemcpyDeviceToDevice, stream);
    hipMemcpyAsync(CAT_IN + UD, GEU, GD * 4, hipMemcpyDeviceToDevice, stream);
    hipMemsetAsync(CAT_OUT, 0, (UD + GD) * 4, stream);
    spmm64_kernel<<<NNZ_UG, 64, 0, stream>>>(ug_rows, ug_cols, ug_vals, CAT_IN, CAT_OUT);
    hipMemcpyAsync(UGP, CAT_OUT,      UD * 4, hipMemcpyDeviceToDevice, stream);
    hipMemcpyAsync(GEU, CAT_OUT + UD, GD * 4, hipMemcpyDeviceToDevice, stream);
    add_f32_kernel<<<((int)UD + 255) / 256, 256, 0, stream>>>(UGP_S, UGP, (int)UD);
    add_f32_kernel<<<((int)GD + 255) / 256, 256, 0, stream>>>(GEU_S, GEU, (int)GD);

    // ui = spmm(ui, concat(uip, ieu))
    hipMemcpyAsync(CAT_IN,      UIP, UD * 4, hipMemcpyDeviceToDevice, stream);
    hipMemcpyAsync(CAT_IN + UD, IEU, ID * 4, hipMemcpyDeviceToDevice, stream);
    hipMemsetAsync(CAT_OUT, 0, (UD + ID) * 4, stream);
    spmm64_kernel<<<NNZ_UI, 64, 0, stream>>>(ui_rows, ui_cols, ui_vals, CAT_IN, CAT_OUT);
    hipMemcpyAsync(UIP, CAT_OUT,      UD * 4, hipMemcpyDeviceToDevice, stream);
    hipMemcpyAsync(IEU, CAT_OUT + UD, ID * 4, hipMemcpyDeviceToDevice, stream);
    add_f32_kernel<<<((int)UD + 255) / 256, 256, 0, stream>>>(UIP_S, UIP, (int)UD);

    // gi = spmm(gi, concat(gei, ieg))
    hipMemcpyAsync(CAT_IN,      GEI, GD * 4, hipMemcpyDeviceToDevice, stream);
    hipMemcpyAsync(CAT_IN + GD, IEG, ID * 4, hipMemcpyDeviceToDevice, stream);
    hipMemsetAsync(CAT_OUT, 0, (GD + ID) * 4, stream);
    spmm64_kernel<<<NNZ_GI, 64, 0, stream>>>(gi_rows, gi_cols, gi_vals, CAT_IN, CAT_OUT);
    hipMemcpyAsync(GEI, CAT_OUT,      GD * 4, hipMemcpyDeviceToDevice, stream);
    hipMemcpyAsync(IEG, CAT_OUT + GD, ID * 4, hipMemcpyDeviceToDevice, stream);
    add_f32_kernel<<<((int)GD + 255) / 256, 256, 0, stream>>>(GEI_S, GEI, (int)GD);
  }

  // ---- phase 5: outputs (means = sums / (NLAYER+1)) ----
  const float inv3 = 1.0f / (float)(NLAYER + 1);
  size_t o = 0;
  gather_concat_kernel<<<BATCH, 128, 0, stream>>>(user_inputs, UGP_S, UIP_S, out + o, inv3);
  o += (size_t)BATCH * 128;
  gather_concat_kernel<<<BATCH, 128, 0, stream>>>(pos_groups, GEU_S, GEI_S, out + o, inv3);
  o += (size_t)BATCH * 128;
  gather_concat_kernel<<<BATCH, 128, 0, stream>>>(neg_groups, GEU_S, GEI_S, out + o, inv3);
  o += (size_t)BATCH * 128;
  gather_kernel<<<BATCH, 64, 0, stream>>>(user_inputs, user_emb, out + o);
  o += (size_t)BATCH * 64;
  gather_kernel<<<BATCH, 64, 0, stream>>>(pos_groups, group_emb, out + o);
  o += (size_t)BATCH * 64;
  gather_kernel<<<BATCH, 64, 0, stream>>>(neg_groups, group_emb, out + o);
}
